// TileWiseFloat8GroupedLinear_32023276159269
// MI455X (gfx1250) — 8-bit matrix operand order repaired and checked by reading, compiled, not run on hardware
//
#include <hip/hip_runtime.h>
#include <hip/hip_bf16.h>


typedef __attribute__((ext_vector_type(16))) int   v16i;
typedef __attribute__((ext_vector_type(8)))  float v8f;
typedef __attribute__((ext_vector_type(4)))  float v4f;

#define SEQ   8192
#define DIN   2048
#define DOUT  2048
#define NE    8
#define G     128
#define KG    (DIN / G)
#define NB    (DOUT / G)
#define FP8_MAXV 448.0f
#define EPSV     1e-12f

__global__ void quant_x_kernel(const float* __restrict__ x,
                               unsigned char* __restrict__ xq,
                               float* __restrict__ xs) {
    const int chunk = (blockIdx.x * blockDim.x + threadIdx.x) >> 5;
    const int lane  = threadIdx.x & 31;
    const int row   = chunk / KG;
    const int kg    = chunk % KG;

    const v4f v = __builtin_nontemporal_load(
        ((const v4f*)(x + (size_t)row * DIN + kg * G)) + lane);
    float am = fmaxf(fmaxf(fabsf(v.x), fabsf(v.y)), fmaxf(fabsf(v.z), fabsf(v.w)));
#pragma unroll
    for (int off = 16; off > 0; off >>= 1)
        am = fmaxf(am, __shfl_xor(am, off, 32));

    const float scale = fmaxf(am, EPSV) * (1.0f / FP8_MAXV);
    const float inv   = 1.0f / scale;

    int pk = __builtin_amdgcn_cvt_pk_fp8_f32(v.x * inv, v.y * inv, 0,  false);
    pk     = __builtin_amdgcn_cvt_pk_fp8_f32(v.z * inv, v.w * inv, pk, true);
    ((int*)(xq + (size_t)row * DIN + kg * G))[lane] = pk;
    if (lane == 0) xs[row * KG + kg] = scale;
}

__global__ void quant_w_kernel(const float* __restrict__ w,
                               unsigned char* __restrict__ wq,
                               float* __restrict__ wsc) {
    const int kb = blockIdx.x;
    const int nb = blockIdx.y;
    const int e  = blockIdx.z;

    const int t  = threadIdx.x;
    const int r  = t >> 1;
    const int c0 = (t & 1) * 64;

    const v4f* src = (const v4f*)(w + ((size_t)e * DOUT + nb * G + r) * DIN
                                    + kb * G + c0);
    v4f   regs[16];
    float am = 0.0f;
#pragma unroll
    for (int i = 0; i < 16; ++i) {
        v4f v = __builtin_nontemporal_load(src + i);
        regs[i] = v;
        am = fmaxf(am, fmaxf(fmaxf(fabsf(v.x), fabsf(v.y)),
                             fmaxf(fabsf(v.z), fabsf(v.w))));
    }
#pragma unroll
    for (int off = 16; off > 0; off >>= 1)
        am = fmaxf(am, __shfl_xor(am, off, 32));

    __shared__ float red[8];
    if ((t & 31) == 0) red[t >> 5] = am;
    __syncthreads();
    am = red[0];
#pragma unroll
    for (int i = 1; i < 8; ++i) am = fmaxf(am, red[i]);

    const float scale = fmaxf(am, EPSV) * (1.0f / FP8_MAXV);
    const float inv   = 1.0f / scale;

    int* dst = (int*)(wq + ((size_t)e * DOUT + nb * G + r) * DIN + kb * G + c0);
#pragma unroll
    for (int i = 0; i < 16; ++i) {
        int pk = __builtin_amdgcn_cvt_pk_fp8_f32(regs[i].x * inv, regs[i].y * inv, 0,  false);
        pk     = __builtin_amdgcn_cvt_pk_fp8_f32(regs[i].z * inv, regs[i].w * inv, pk, true);
        dst[i] = pk;
    }
    if (t == 0) wsc[(e * NB + nb) * KG + kb] = scale;
}

__global__ void __launch_bounds__(256)
gemm_fp8_kernel(const unsigned char* __restrict__ xq,
                const float* __restrict__ xs,
                const unsigned char* __restrict__ wq,
                const float* __restrict__ wsc,
                const int* __restrict__ tpe,
                float* __restrict__ out) {
    const int lane  = threadIdx.x & 31;
    const int wave  = threadIdx.x >> 5;
    const int waveM = wave >> 1;
    const int waveN = wave & 1;
    const int half  = lane >> 4;
    const int l15   = lane & 15;

    const int m_base = blockIdx.y * 128 + waveM * 32;
    const int n_base = blockIdx.x * 128 + waveN * 64;
    const int nb     = blockIdx.x;

    int e = -1, st = 0, en = 0, s = 0;
#pragma unroll
    for (int i = 0; i < NE; ++i) {
        int c = tpe[i];
        if (e < 0 && m_base >= s && m_base < s + c) { e = i; st = s; en = s + c; }
        s += c;
    }
    if (e < 0) { e = 0; st = 0; en = 0; }

    const int  tok0   = m_base + l15;
    const int  tok1   = m_base + 16 + l15;
    const bool valid0 = (tok0 >= st) && (tok0 < en);
    const bool valid1 = (tok1 >= st) && (tok1 < en);

    const unsigned char* arow0 = xq + (size_t)tok0 * DIN + 16 * half;
    const unsigned char* arow1 = xq + (size_t)tok1 * DIN + 16 * half;
    const unsigned char* bcol  = wq + ((size_t)e * DOUT + n_base + l15) * DIN
                                    + 16 * half;
    const float* xsrow = xs + (size_t)(m_base + lane) * KG;
    const float* wsrow = wsc + (e * NB + nb) * KG;

    float acc[2][4][8];
#pragma unroll
    for (int tm = 0; tm < 2; ++tm)
#pragma unroll
        for (int tn = 0; tn < 4; ++tn)
#pragma unroll
            for (int i = 0; i < 8; ++i) acc[tm][tn][i] = 0.0f;

    for (int kb = 0; kb < KG; ++kb) {
        v16i a0 = {}, a1 = {};
        if (valid0) {
#pragma unroll
            for (int j = 0; j < 8; ++j) {
                int2 d = *(const int2*)(arow0 + kb * G + 32 * (j >> 1) + 8 * (j & 1));
                a0[2 * j] = d.x;  a0[2 * j + 1] = d.y;
            }
        }
        if (valid1) {
#pragma unroll
            for (int j = 0; j < 8; ++j) {
                int2 d = *(const int2*)(arow1 + kb * G + 32 * (j >> 1) + 8 * (j & 1));
                a1[2 * j] = d.x;  a1[2 * j + 1] = d.y;
            }
        }
        if (kb + 1 < KG) {
            __builtin_prefetch(arow0 + (kb + 1) * G, 0, 1);
            __builtin_prefetch(arow1 + (kb + 1) * G, 0, 1);
        }

        const float sb   = wsrow[kb];
        const float myxs = xsrow[kb];
        float sab[2][8];
#pragma unroll
        for (int tm = 0; tm < 2; ++tm)
#pragma unroll
            for (int i = 0; i < 8; ++i)
                sab[tm][i] = __shfl(myxs, 16 * tm + i + 8 * half, 32) * sb;

#pragma unroll
        for (int tn = 0; tn < 4; ++tn) {
            v16i b;
            const unsigned char* bp = bcol + (size_t)(16 * tn) * DIN + kb * G;
#pragma unroll
            for (int g = 0; g < 4; ++g) {
                int4 d = *(const int4*)(bp + 32 * g);
                b[4 * g]     = d.x;  b[4 * g + 1] = d.y;
                b[4 * g + 2] = d.z;  b[4 * g + 3] = d.w;
            }
            v8f cz = {};
            v8f d0 = __builtin_amdgcn_wmma_f32_16x16x128_fp8_fp8(
                         a0, b, (short)0, cz, false, false);
            v8f d1 = __builtin_amdgcn_wmma_f32_16x16x128_fp8_fp8(
                         a1, b, (short)0, cz, false, false);
#pragma unroll
            for (int i = 0; i < 8; ++i) {
                acc[0][tn][i] = fmaf(d0[i], sab[0][i], acc[0][tn][i]);
                acc[1][tn][i] = fmaf(d1[i], sab[1][i], acc[1][tn][i]);
            }
        }
    }

#pragma unroll
    for (int tm = 0; tm < 2; ++tm)
#pragma unroll
        for (int tn = 0; tn < 4; ++tn)
#pragma unroll
            for (int i = 0; i < 8; ++i) {
                const int row = m_base + 16 * tm + i + 8 * half;
                const int col = n_base + 16 * tn + l15;
                __builtin_nontemporal_store(acc[tm][tn][i],
                                            out + (size_t)row * DOUT + col);
            }
}

extern "C" void kernel_launch(void* const* d_in, const int* in_sizes, int n_in,
                              void* d_out, int out_size, void* d_ws, size_t ws_size,
                              hipStream_t stream) {
    const float* x   = (const float*)d_in[0];
    const float* w   = (const float*)d_in[1];
    const int*   tpe = (const int*)d_in[2];
    float*       out = (float*)d_out;

    unsigned char* ws8 = (unsigned char*)d_ws;
    unsigned char* xq  = ws8;
    unsigned char* wq  = ws8 + (size_t)SEQ * DIN;
    float*         xs  = (float*)(ws8 + (size_t)SEQ * DIN + (size_t)NE * DOUT * DIN);
    float*         wsc = xs + (size_t)SEQ * KG;

    {
        const int chunks  = SEQ * KG;
        const int threads = 256;
        const int blocks  = chunks * 32 / threads;
        quant_x_kernel<<<blocks, threads, 0, stream>>>(x, xq, xs);
    }
    {
        dim3 grid(KG, NB, NE);
        quant_w_kernel<<<grid, 256, 0, stream>>>(w, wq, wsc);
    }
    {
        dim3 grid(DOUT / 128, SEQ / 128);
        gemm_fp8_kernel<<<grid, 256, 0, stream>>>(xq, xs, wq, wsc, tpe, out);
    }
}
